// Convolution_72335839200037
// MI455X (gfx1250) — matrix operand order repaired and checked by reading, not compiled, not run on hardware
//
#include <hip/hip_runtime.h>

typedef __attribute__((ext_vector_type(16))) _Float16 v16h;
typedef __attribute__((ext_vector_type(8)))  float    v8f;

#define DEVINL __device__ __forceinline__

// ---------------------------------------------------------------------------
// WMMA helpers (CDNA5 v_wmma_f32_16x16x32_f16, wave32)
// ---------------------------------------------------------------------------
DEVINL v8f wmma_f16(v16h a, v16h b, v8f c) {
  return __builtin_amdgcn_wmma_f32_16x16x32_f16(
      /*neg_a=*/false, a, /*neg_b=*/false, b,
      /*c_mod=*/(short)0, c, /*reuse_a=*/false, /*reuse_b=*/false);
}

DEVINL v8f v8f_zero() {
  v8f c;
#pragma unroll
  for (int r = 0; r < 8; ++r) c[r] = 0.f;
  return c;
}

// A-frag (16x32 f16): lane holds row m=lane&15; K = {kb8..kb8+7, 16+kb8..16+kb8+7}
DEVINL v16h load_a_f32(const float* __restrict__ rowPtr, int colBase, int stride, int kb8) {
  v16h a;
#pragma unroll
  for (int j = 0; j < 8; ++j) {
    a[j]     = (_Float16)rowPtr[colBase + (kb8 + j) * stride];
    a[j + 8] = (_Float16)rowPtr[colBase + (16 + kb8 + j) * stride];
  }
  return a;
}

// A-frag from LDS f16 row (64 contiguous halves per matrix row)
DEVINL v16h load_a_lds(const _Float16* __restrict__ row, int kBase, int kb8) {
  v16h a;
#pragma unroll
  for (int j = 0; j < 8; ++j) {
    a[j]     = row[kBase + kb8 + j];
    a[j + 8] = row[kBase + 16 + kb8 + j];
  }
  return a;
}

// B-frag (32x16 f16): lane holds col n; K = kBase + {kb8..kb8+7, 16+kb8..16+kb8+7}, kb8 = kb16/2  (W row-major [K][ncols])
DEVINL v16h load_b_f32(const float* __restrict__ W, int ncols, int kBase, int kb16,
                       int n, float scale) {
  v16h b;
#pragma unroll
  for (int j = 0; j < 16; ++j)
    b[j] = (_Float16)(W[(size_t)(kBase + (kb16 >> 1) + (j & 7) + ((j >> 3) << 4)) * ncols + n] * scale);
  return b;
}

// relaxed agent-scope f32 atomic add -> global_atomic_add_f32 (no return)
DEVINL void atomF(float* p, float v) {
  __hip_atomic_fetch_add(p, v, __ATOMIC_RELAXED, __HIP_MEMORY_SCOPE_AGENT);
}

DEVINL int uniform_wave() {               // wave id, pinned to SGPR
  return __builtin_amdgcn_readfirstlane((int)(threadIdx.x >> 5));
}

// normalization constants
#define RS8  0.35355339059327373f   // 1/sqrt(8)
#define RS32 0.17677669529663687f   // 1/sqrt(32)
#define RS64 0.125f                 // 1/sqrt(64)
#define RS96 0.10206207261596575f   // 1/sqrt(96)
#define S10  0.31622776601683794f   // 1/sqrt(10)
#define INV3 0.5773502691896258f    // 1/sqrt(3)

// ---------------------------------------------------------------------------
// Kernel 1: per-node pre-linears (sc = lin(W_sc), f = lin(W1)), WMMA tiles.
// ---------------------------------------------------------------------------
__global__ __launch_bounds__(256) void node_pre_kernel(
    const float* __restrict__ ni, const float* __restrict__ attr,
    const float* __restrict__ Wscs, const float* __restrict__ Wscv,
    const float* __restrict__ W1s,  const float* __restrict__ W1v,
    float* __restrict__ fs, float* __restrict__ fv, float* __restrict__ sc)
{
  const int lane = threadIdx.x & 31;
  const int wave = uniform_wave();
  const int rowBase = blockIdx.x * 16;
  const int m    = lane & 15;
  const int kb8  = (lane >> 4) * 8;
  const int kb16 = (lane >> 4) * 16;
  const float* rowPtr = ni + (size_t)(rowBase + m) * 160;
  const float* attrB  = attr + rowBase;

  for (int t = wave; t < 20; t += 8) {
    v8f c = v8f_zero();
    float* dstBase;     // wave-uniform destination select (scalar branch)
    int    rstride, col;
    if (t < 8) {                       // scalar irreps: (16x64)@(64x64)
      const float* W = (t < 4) ? W1s : Wscs;
      const int ct = t & 3;
      const int n  = ct * 16 + m;
#pragma unroll
      for (int kk = 0; kk < 2; ++kk) {
        v16h a = load_a_f32(rowPtr, kk * 32, 1, kb8);
        v16h b = load_b_f32(W, 64, kk * 32, kb16, n, RS64);
        c = wmma_f16(a, b, c);
      }
      if (t < 4) { dstBase = fs; rstride = 64;  col = n; }
      else       { dstBase = sc; rstride = 160; col = n; }
    } else {                           // vector irreps: per-d (16x32)@(32x32)
      const int idx = t - 8;
      const bool isF = idx < 6;
      const int id2 = isF ? idx : idx - 6;
      const int d = id2 >> 1, ct = id2 & 1;
      const float* W = isF ? W1v : Wscv;
      const int n = ct * 16 + m;
      v16h a = load_a_f32(rowPtr, 64 + d, 3, kb8);
      v16h b = load_b_f32(W, 32, 0, kb16, n, RS32);
      c = wmma_f16(a, b, c);
      if (isF) { dstBase = fv; rstride = 96;  col = n * 3 + d; }
      else     { dstBase = sc; rstride = 160; col = 64 + n * 3 + d; }
    }
#pragma unroll
    for (int r = 0; r < 8; ++r) {
      const int mr = (lane >> 4) * 8 + r;
      dstBase[(size_t)(rowBase + mr) * rstride + col] = c[r] * attrB[mr];
    }
  }
}

// ---------------------------------------------------------------------------
// Kernel 2: fully fused edge pipeline. 4 waves/block, 16 edges per wave.
// ---------------------------------------------------------------------------
__global__ __launch_bounds__(128) void edge_kernel(
    const float* __restrict__ es, const float* __restrict__ ea,
    const int* __restrict__ esrc, const int* __restrict__ edst,
    const float* __restrict__ fcW1, const float* __restrict__ fcW2,
    const float* __restrict__ fs, const float* __restrict__ fv,
    float* __restrict__ agg, int nE)
{
  __shared__ _Float16 hT[4][16 * 64];    //  8 KB
  __shared__ float    wT[4][16 * 192];   // 48 KB

  const int lane = threadIdx.x & 31;
  const int wave = uniform_wave();
  const int m    = lane & 15;
  const int kb8  = (lane >> 4) * 8;
  const int kb16 = (lane >> 4) * 16;
  const int e0   = (blockIdx.x * 4 + wave) * 16;

  if (e0 + 64 < nE) __builtin_prefetch(es + (size_t)(e0 + 64) * 8, 0, 0);

  // ---- layer 1: h = silu( es @ fcW1 * 1/sqrt(8) ), K=8 zero-padded to 32 ----
  {
    v16h a;
    const float* rp = es + (size_t)(e0 + m) * 8;
#pragma unroll
    for (int j = 0; j < 8; ++j) {
      int k = kb8 + j;
      a[j]     = (k < 8) ? (_Float16)rp[k] : (_Float16)0.f;
      a[j + 8] = (_Float16)0.f;
    }
#pragma unroll
    for (int ct = 0; ct < 4; ++ct) {
      const int n = ct * 16 + m;
      v16h b;
#pragma unroll
      for (int j = 0; j < 16; ++j) {
        int k = kb16 + j;
        b[j] = (k < 8) ? (_Float16)(fcW1[k * 64 + n] * RS8) : (_Float16)0.f;
      }
      v8f c = wmma_f16(a, b, v8f_zero());
#pragma unroll
      for (int r = 0; r < 8; ++r) {
        int mr = (lane >> 4) * 8 + r;
        float x = c[r];
        hT[wave][mr * 64 + n] = (_Float16)(x / (1.f + __expf(-x)));   // SiLU
      }
    }
  }
  asm volatile("s_wait_dscnt 0" ::: "memory");

  // ---- layer 2: w = h @ fcW2 * 1/sqrt(64)  (16x64)@(64x192) ----
  {
    v16h a0 = load_a_lds(&hT[wave][m * 64], 0,  kb8);
    v16h a1 = load_a_lds(&hT[wave][m * 64], 32, kb8);
#pragma unroll
    for (int ct = 0; ct < 12; ++ct) {
      const int n = ct * 16 + m;
      v8f c = v8f_zero();
      c = wmma_f16(a0, load_b_f32(fcW2, 192, 0,  kb16, n, RS64), c);
      c = wmma_f16(a1, load_b_f32(fcW2, 192, 32, kb16, n, RS64), c);
#pragma unroll
      for (int r = 0; r < 8; ++r)
        wT[wave][((lane >> 4) * 8 + r) * 192 + n] = c[r];
    }
  }
  asm volatile("s_wait_dscnt 0" ::: "memory");

  // ---- per-edge tensor products + atomic segment-sum (2 lanes per edge) ----
  const int el   = lane & 15;
  const int part = lane >> 4;
  const int e = e0 + el;
  const int src = esrc[e];
  const int dst = edst[e];
  const float y0  = ea[(size_t)e * 4 + 0];
  const float y1x = ea[(size_t)e * 4 + 1];
  const float y1y = ea[(size_t)e * 4 + 2];
  const float y1z = ea[(size_t)e * 4 + 3];
  const float* xs = fs + (size_t)src * 64;
  const float* xv = fv + (size_t)src * 96;
  float* ag = agg + (size_t)dst * 384;
  const float* wr = &wT[wave][el * 192];     // cols: [A64 | B64 | C32 | D32]

#pragma unroll 4
  for (int u = part * 32; u < part * 32 + 32; ++u) {
    float x = xs[u];
    atomF(&ag[u], wr[u] * x * y0 * S10);                       // outA
    float bx = wr[64 + u] * x * S10;                           // outB
    atomF(&ag[96 + u * 3 + 0], bx * y1x);
    atomF(&ag[96 + u * 3 + 1], bx * y1y);
    atomF(&ag[96 + u * 3 + 2], bx * y1z);
  }
#pragma unroll 4
  for (int u = part * 16; u < part * 16 + 16; ++u) {
    float vx = xv[u * 3 + 0], vy = xv[u * 3 + 1], vz = xv[u * 3 + 2];
    atomF(&ag[64 + u], wr[160 + u] * (vx * y1x + vy * y1y + vz * y1z) * (INV3 * S10)); // outD
    float cy = wr[128 + u] * y0 * S10;                         // outC
    atomF(&ag[288 + u * 3 + 0], cy * vx);
    atomF(&ag[288 + u * 3 + 1], cy * vy);
    atomF(&ag[288 + u * 3 + 2], cy * vz);
  }
}

// ---------------------------------------------------------------------------
// Kernel 3: post-linear (W2, K=96) + angle + cos/sin mixing. WMMA tiles.
// ---------------------------------------------------------------------------
__global__ __launch_bounds__(256) void node_post_kernel(
    const float* __restrict__ agg, const float* __restrict__ attr,
    const float* __restrict__ W2s, const float* __restrict__ W2v,
    const float* __restrict__ W3,  const float* __restrict__ sc,
    float* __restrict__ out)
{
  __shared__ float conv[16 * 160];
  __shared__ float cosA[16], sinA[16];

  const int lane = threadIdx.x & 31;
  const int wave = uniform_wave();
  const int rowBase = blockIdx.x * 16;
  const int m    = lane & 15;
  const int kb8  = (lane >> 4) * 8;
  const int kb16 = (lane >> 4) * 16;
  const float* attrB = attr + rowBase;

  for (int t = wave; t < 10; t += 8) {
    v8f c = v8f_zero();
    int col;
    if (t < 4) {                       // c_s = ms(16x96) @ W2_s(96x64)
      const int n = t * 16 + m;
      const float* rowPtr = agg + (size_t)(rowBase + m) * 384;
#pragma unroll
      for (int kk = 0; kk < 3; ++kk) {
        v16h a = load_a_f32(rowPtr, kk * 32, 1, kb8);
        v16h b = load_b_f32(W2s, 64, kk * 32, kb16, n, RS96);
        c = wmma_f16(a, b, c);
      }
      col = n;
    } else {                           // c_v[:,:,d] = mv_d(16x96) @ W2_v(96x32)
      const int idx = t - 4;
      const int d = idx >> 1, ct = idx & 1;
      const int n = ct * 16 + m;
      const float* rowPtr = agg + (size_t)(rowBase + m) * 384 + 96 + d;
#pragma unroll
      for (int kk = 0; kk < 3; ++kk) {
        v16h a = load_a_f32(rowPtr, kk * 96, 3, kb8);   // col = 96 + k*3 + d
        v16h b = load_b_f32(W2v, 32, kk * 32, kb16, n, RS96);
        c = wmma_f16(a, b, c);
      }
      col = 64 + n * 3 + d;
    }
#pragma unroll
    for (int r = 0; r < 8; ++r) {
      const int mr = (lane >> 4) * 8 + r;
      conv[mr * 160 + col] = c[r] * attrB[mr];
    }
  }
  __syncthreads();

  if (threadIdx.x < 16) {              // angle = 0.1 * (ms @ W3) * attr / sqrt(96)
    int row = rowBase + threadIdx.x;
    const float* ar = agg + (size_t)row * 384;
    float s = 0.f;
#pragma unroll 8
    for (int k = 0; k < 96; ++k) s += ar[k] * W3[k];
    float ang = 0.1f * s * RS96 * attr[row];
    cosA[threadIdx.x] = __cosf(ang);
    sinA[threadIdx.x] = __sinf(ang);
  }
  __syncthreads();

  for (int i = threadIdx.x; i < 16 * 160; i += 256) {
    int r = i / 160, cc = i % 160;
    size_t row = (size_t)(rowBase + r);
    out[row * 160 + cc] = cosA[r] * sc[row * 160 + cc] + sinA[r] * conv[r * 160 + cc];
  }
}

// ---------------------------------------------------------------------------
// Launch. Workspace: f_s(N*64) | f_v(N*96) | sc(N*160) | agg(N*384)  (~108 MB)
// ---------------------------------------------------------------------------
extern "C" void kernel_launch(void* const* d_in, const int* in_sizes, int n_in,
                              void* d_out, int out_size, void* d_ws, size_t ws_size,
                              hipStream_t stream) {
  const float* ni   = (const float*)d_in[0];
  const float* attr = (const float*)d_in[1];
  const float* ea   = (const float*)d_in[2];
  const float* es   = (const float*)d_in[3];
  const float* Wscs = (const float*)d_in[4];
  const float* Wscv = (const float*)d_in[5];
  const float* W1s  = (const float*)d_in[6];
  const float* W1v  = (const float*)d_in[7];
  const float* W2s  = (const float*)d_in[8];
  const float* W2v  = (const float*)d_in[9];
  const float* W3   = (const float*)d_in[10];
  const float* fcW1 = (const float*)d_in[11];
  const float* fcW2 = (const float*)d_in[12];
  const int* esrc   = (const int*)d_in[13];
  const int* edst   = (const int*)d_in[14];
  float* out = (float*)d_out;

  const int nN = in_sizes[0] / 160;   // 40000 (divisible by 16)
  const int nE = in_sizes[13];        // 400000 (divisible by 64)

  float* fs  = (float*)d_ws;
  float* fv  = fs + (size_t)nN * 64;
  float* sc  = fv + (size_t)nN * 96;
  float* agg = sc + (size_t)nN * 160;

  hipMemsetAsync(agg, 0, (size_t)nN * 384 * sizeof(float), stream);
  node_pre_kernel<<<nN / 16, 256, 0, stream>>>(ni, attr, Wscs, Wscv, W1s, W1v, fs, fv, sc);
  edge_kernel<<<nE / 64, 128, 0, stream>>>(es, ea, esrc, edst, fcW1, fcW2, fs, fv, agg, nE);
  node_post_kernel<<<nN / 16, 256, 0, stream>>>(agg, attr, W2s, W2v, W3, sc, out);
}
